// RoPEMultiheadAttention_32942399160838
// MI455X (gfx1250) — matrix operand order repaired and checked by reading, not compiled, not run on hardware
//
#include <hip/hip_runtime.h>

// ---------- types ----------
typedef __attribute__((ext_vector_type(16))) __bf16 v16bf;
typedef __attribute__((ext_vector_type(8)))  float  v8f;

union FragU {
    v16bf v;
    uint4 q[2];
    unsigned short u[16];
};

// ---------- bf16 helpers (bit-level, round-to-nearest-even) ----------
__device__ __forceinline__ unsigned short f2bf(float f) {
    unsigned int u = __float_as_uint(f);
    unsigned int r = u + 0x7FFFu + ((u >> 16) & 1u);
    return (unsigned short)(r >> 16);
}
__device__ __forceinline__ float bf2f(unsigned short h) {
    return __uint_as_float(((unsigned int)h) << 16);
}

// ---------- constants ----------
#define BATCH   2
#define SEQ     2048
#define DMODEL  1024
#define HEADS   16
#define HDIM    64
#define THREED  3072
#define BT      4096   // BATCH*SEQ

// ---------- 1) fp32 -> bf16 convert ----------
__global__ void cvt_f32_bf16(const float* __restrict__ in,
                             unsigned short* __restrict__ out, int n) {
    int i = blockIdx.x * blockDim.x + threadIdx.x;
    if (i < n) out[i] = f2bf(in[i]);
}

// ---------- 2/5) bf16 GEMM:  C[M,N] = A[M,K] * W[N,K]^T + bias ----------
// One wave computes a 32x64 output tile: 2 A-frags x 4 B-frags -> 8 WMMAs/k-step.
__global__ __launch_bounds__(32)
void gemm_bf16_wmma(const unsigned short* __restrict__ A,
                    const unsigned short* __restrict__ W,
                    const float* __restrict__ bias,
                    unsigned short* __restrict__ outb,
                    float* __restrict__ outf,
                    int M, int N, int K, int f32out) {
    const int lane = threadIdx.x;
    const int hf   = lane >> 4;     // wave half
    const int lr   = lane & 15;
    const int n0   = blockIdx.x * 64;
    const int m0   = blockIdx.y * 32;

    v8f acc[2][4];
    const v8f zero = {0.f,0.f,0.f,0.f,0.f,0.f,0.f,0.f};
    for (int mt = 0; mt < 2; ++mt)
        for (int t = 0; t < 4; ++t) acc[mt][t] = zero;

    const unsigned short* arow0 = A + (size_t)(m0 + lr) * K;
    const unsigned short* arow1 = A + (size_t)(m0 + 16 + lr) * K;

    for (int k0 = 0; k0 < K; k0 += 32) {
        // A fragments (16x32, MxK): lane lr = row, K elems {0..7,16..23}+8*hf
        FragU a0, a1;
        a0.q[0] = *(const uint4*)(arow0 + k0 + hf * 8);
        a0.q[1] = *(const uint4*)(arow0 + k0 + 16 + hf * 8);
        a1.q[0] = *(const uint4*)(arow1 + k0 + hf * 8);
        a1.q[1] = *(const uint4*)(arow1 + k0 + 16 + hf * 8);
        for (int t = 0; t < 4; ++t) {
            // B fragment (32x16, KxN): column = weight row n; K elems {0..7,16..23}+8*hf
            const unsigned short* wrow =
                W + (size_t)(n0 + t * 16 + lr) * K + k0 + hf * 8;
            FragU b;
            b.q[0] = ((const uint4*)wrow)[0];
            b.q[1] = ((const uint4*)wrow)[2];
            acc[0][t] = __builtin_amdgcn_wmma_f32_16x16x32_bf16(
                false, a0.v, false, b.v, (short)0, acc[0][t], false, false);
            acc[1][t] = __builtin_amdgcn_wmma_f32_16x16x32_bf16(
                false, a1.v, false, b.v, (short)0, acc[1][t], false, false);
        }
    }

    for (int mt = 0; mt < 2; ++mt)
        for (int t = 0; t < 4; ++t) {
            const int n = n0 + t * 16 + lr;
            const float bv = bias[n];
            for (int j = 0; j < 8; ++j) {
                const float val = acc[mt][t][j] + bv;
                const size_t idx = (size_t)(m0 + mt * 16 + j + 8 * hf) * N + n;
                if (f32out) outf[idx] = val;
                else        outb[idx] = f2bf(val);
            }
        }
}

// ---------- 3) RoPE + relayout: q,k -> [B,H,T,64], v -> [B,H,64,T] (transposed) ----------
__global__ void rope_kernel(const unsigned short* __restrict__ qkv,
                            unsigned short* __restrict__ Qr,
                            unsigned short* __restrict__ Kr,
                            unsigned short* __restrict__ Vt) {
    const int idx = blockIdx.x * blockDim.x + threadIdx.x;
    const int d2 = idx & 31;
    const int h  = (idx >> 5) & 15;
    const int t  = (idx >> 9) & 2047;
    const int b  = idx >> 20;
    if (b >= BATCH) return;

    const size_t bt = (size_t)b * SEQ + t;
    const int bh = b * HEADS + h;
    const unsigned short* qb = qkv + bt * THREED + h * HDIM;

    const float q1 = bf2f(qb[2 * d2]),        q2 = bf2f(qb[2 * d2 + 1]);
    const float k1 = bf2f(qb[1024 + 2 * d2]), k2 = bf2f(qb[1024 + 2 * d2 + 1]);

    const float inv = __powf(10000.0f, -(float)d2 * (1.0f / 32.0f));
    const float ang = (float)t * inv;
    const float sn = __sinf(ang), cs = __cosf(ang);

    const size_t ob = ((size_t)bh * SEQ + t) * HDIM;
    Qr[ob + d2]      = f2bf(q1 * cs - q2 * sn);
    Qr[ob + 32 + d2] = f2bf(q1 * sn + q2 * cs);
    Kr[ob + d2]      = f2bf(k1 * cs - k2 * sn);
    Kr[ob + 32 + d2] = f2bf(k1 * sn + k2 * cs);
    // V transposed per head: Vt[bh][d][t]
    Vt[((size_t)bh * HDIM + 2 * d2)     * SEQ + t] = qb[2048 + 2 * d2];
    Vt[((size_t)bh * HDIM + 2 * d2 + 1) * SEQ + t] = qb[2048 + 2 * d2 + 1];
}

// ---------- 4) flash attention: one wave per (b,h, 32-query tile) ----------
// K/V fragments loaded once per 32-key step, reused by both 16-row query tiles.
__global__ __launch_bounds__(32)
void attn_kernel(const unsigned short* __restrict__ Q,
                 const unsigned short* __restrict__ Km,
                 const unsigned short* __restrict__ Vt,
                 unsigned short* __restrict__ Obuf) {
    const int lane = threadIdx.x;
    const int hf   = lane >> 4;
    const int lr   = lane & 15;
    const int bh   = blockIdx.y;          // b*HEADS + h
    const int q0   = blockIdx.x * 32;
    const int b    = bh >> 4;
    const int h    = bh & 15;
    const float scale = 0.125f;           // 1/sqrt(64)

    __shared__ __align__(16) unsigned short pbuf[16 * 32];

    // Q tiles (2 x 16x64) -> 4 resident A-fragments
    FragU aq[2][2];
    for (int mt = 0; mt < 2; ++mt) {
        const unsigned short* qrow =
            Q + ((size_t)bh * SEQ + q0 + mt * 16 + lr) * HDIM;
        for (int kc = 0; kc < 2; ++kc) {
            aq[mt][kc].q[0] = *(const uint4*)(qrow + kc * 32 + hf * 8);
            aq[mt][kc].q[1] = *(const uint4*)(qrow + kc * 32 + 16 + hf * 8);
        }
    }

    v8f o[2][4];
    const v8f zero = {0.f,0.f,0.f,0.f,0.f,0.f,0.f,0.f};
    for (int mt = 0; mt < 2; ++mt)
        for (int t = 0; t < 4; ++t) o[mt][t] = zero;
    float mrow[2][8], lsum[2][8];
    for (int mt = 0; mt < 2; ++mt)
        for (int j = 0; j < 8; ++j) { mrow[mt][j] = -3.0e38f; lsum[mt][j] = 0.f; }

    for (int s0 = 0; s0 < SEQ; s0 += 32) {
        // ---- hoist K fragments (4) and V fragments (4) for this key block ----
        FragU kf[2][2];
        for (int st = 0; st < 2; ++st) {
            const unsigned short* krow =
                Km + ((size_t)bh * SEQ + s0 + st * 16 + lr) * HDIM;
            for (int kc = 0; kc < 2; ++kc) {
                const uint4* pb = (const uint4*)(krow + kc * 32 + hf * 8);
                kf[st][kc].q[0] = pb[0];
                kf[st][kc].q[1] = pb[2];
            }
        }
        FragU vf[4];
        for (int t = 0; t < 4; ++t) {
            // Vt[bh][d][t]: column d = t*16+lr, keys {0..7,16..23}+8*hf
            const unsigned short* vrow =
                Vt + ((size_t)bh * HDIM + t * 16 + lr) * SEQ + s0 + hf * 8;
            vf[t].q[0] = ((const uint4*)vrow)[0];
            vf[t].q[1] = ((const uint4*)vrow)[2];
        }

        for (int mt = 0; mt < 2; ++mt) {
            // ---- S = scale * Q @ K^T  (16 x 32) ----
            v8f s01[2];
            for (int st = 0; st < 2; ++st) {
                v8f s = zero;
                for (int kc = 0; kc < 2; ++kc)
                    s = __builtin_amdgcn_wmma_f32_16x16x32_bf16(
                        false, aq[mt][kc].v, false, kf[st][kc].v,
                        (short)0, s, false, false);
                s01[st] = s * scale;
            }

            // ---- online softmax: rows j+8*hf live across 16-lane groups ----
            float mx[8];
            for (int j = 0; j < 8; ++j) mx[j] = fmaxf(s01[0][j], s01[1][j]);
            for (int mask = 1; mask <= 8; mask <<= 1)
                for (int j = 0; j < 8; ++j)
                    mx[j] = fmaxf(mx[j], __shfl_xor(mx[j], mask, 32));

            float al[8];
            for (int j = 0; j < 8; ++j) {
                const float nm = fmaxf(mrow[mt][j], mx[j]);
                al[j] = __expf(mrow[mt][j] - nm);
                mrow[mt][j] = nm;
            }

            float p0[8], p1[8], ps[8];
            for (int j = 0; j < 8; ++j) {
                p0[j] = __expf(s01[0][j] - mrow[mt][j]);
                p1[j] = __expf(s01[1][j] - mrow[mt][j]);
                ps[j] = p0[j] + p1[j];
            }
            for (int mask = 1; mask <= 8; mask <<= 1)
                for (int j = 0; j < 8; ++j)
                    ps[j] += __shfl_xor(ps[j], mask, 32);
            for (int j = 0; j < 8; ++j)
                lsum[mt][j] = lsum[mt][j] * al[j] + ps[j];
            for (int t = 0; t < 4; ++t)
                for (int j = 0; j < 8; ++j) o[mt][t][j] *= al[j];

            // ---- redistribute P (D-layout) -> A-layout through LDS ----
            __syncthreads();
            for (int j = 0; j < 8; ++j) {
                pbuf[(j + 8 * hf) * 32 + lr]      = f2bf(p0[j]);
                pbuf[(j + 8 * hf) * 32 + 16 + lr] = f2bf(p1[j]);
            }
            __syncthreads();
            FragU pa;
            pa.q[0] = *(const uint4*)(&pbuf[lr * 32 + hf * 8]);
            pa.q[1] = *(const uint4*)(&pbuf[lr * 32 + 16 + hf * 8]);

            // ---- O += P @ V  (K = 32 keys, N = 64 in 4 tiles) ----
            for (int t = 0; t < 4; ++t)
                o[mt][t] = __builtin_amdgcn_wmma_f32_16x16x32_bf16(
                    false, pa.v, false, vf[t].v, (short)0, o[mt][t], false, false);
        }
    }

    // ---- normalize and store as [B*T, DMODEL] bf16 ----
    for (int mt = 0; mt < 2; ++mt)
        for (int t = 0; t < 4; ++t)
            for (int j = 0; j < 8; ++j) {
                const float val = o[mt][t][j] / lsum[mt][j];
                const size_t row = (size_t)b * SEQ + q0 + mt * 16 + j + 8 * hf;
                Obuf[row * DMODEL + h * HDIM + t * 16 + lr] = f2bf(val);
            }
}

// ---------- host launch ----------
extern "C" void kernel_launch(void* const* d_in, const int* in_sizes, int n_in,
                              void* d_out, int out_size, void* d_ws, size_t ws_size,
                              hipStream_t stream) {
    const float* x     = (const float*)d_in[0];   // [B,T,D]
    const float* w_qkv = (const float*)d_in[1];   // [3D,D]
    const float* b_qkv = (const float*)d_in[2];   // [3D]
    const float* w_out = (const float*)d_in[3];   // [D,D]
    const float* b_out = (const float*)d_in[4];   // [D]
    float* out = (float*)d_out;

    // workspace carve-up (bf16 buffers), 256B aligned
    size_t off = 0;
    auto carve = [&](size_t elems) {
        unsigned short* p = (unsigned short*)((char*)d_ws + off);
        off += ((elems * 2) + 255) & ~(size_t)255;
        return p;
    };
    unsigned short* x_bf     = carve((size_t)BT * DMODEL);       // 8 MB
    unsigned short* w_qkv_bf = carve((size_t)THREED * DMODEL);   // 6 MB
    unsigned short* w_out_bf = carve((size_t)DMODEL * DMODEL);   // 2 MB
    unsigned short* qkv_bf   = carve((size_t)BT * THREED);       // 24 MB
    unsigned short* q_r      = carve((size_t)BT * DMODEL);       // 8 MB
    unsigned short* k_r      = carve((size_t)BT * DMODEL);       // 8 MB
    unsigned short* v_t      = carve((size_t)BT * DMODEL);       // 8 MB (transposed per head)
    unsigned short* attn_bf  = carve((size_t)BT * DMODEL);       // 8 MB

    // 1) convert inputs to bf16
    {
        int n = BT * DMODEL;
        cvt_f32_bf16<<<(n + 255) / 256, 256, 0, stream>>>(x, x_bf, n);
        n = THREED * DMODEL;
        cvt_f32_bf16<<<(n + 255) / 256, 256, 0, stream>>>(w_qkv, w_qkv_bf, n);
        n = DMODEL * DMODEL;
        cvt_f32_bf16<<<(n + 255) / 256, 256, 0, stream>>>(w_out, w_out_bf, n);
    }

    // 2) qkv = x @ w_qkv^T + b_qkv   (bf16 out)
    gemm_bf16_wmma<<<dim3(THREED / 64, BT / 32), 32, 0, stream>>>(
        x_bf, w_qkv_bf, b_qkv, qkv_bf, nullptr, BT, THREED, DMODEL, 0);

    // 3) RoPE + relayout q,k -> [B,H,T,64]; v -> [B,H,64,T]
    {
        const int n = BATCH * SEQ * HEADS * 32;  // one thread per rotation pair
        rope_kernel<<<n / 256, 256, 0, stream>>>(qkv_bf, q_r, k_r, v_t);
    }

    // 4) flash attention -> attn_bf [B*T, D]
    attn_kernel<<<dim3(SEQ / 32, BATCH * HEADS), 32, 0, stream>>>(
        q_r, k_r, v_t, attn_bf);

    // 5) out = attn @ w_out^T + b_out   (fp32 out)
    gemm_bf16_wmma<<<dim3(DMODEL / 64, BT / 32), 32, 0, stream>>>(
        attn_bf, w_out_bf, b_out, nullptr, out, BT, DMODEL, DMODEL, 1);
}
